// BasicAttention_30167850287543
// MI455X (gfx1250) — matrix operand order repaired and checked by reading, not compiled, not run on hardware
//
#include <hip/hip_runtime.h>

typedef __attribute__((ext_vector_type(16))) __bf16 v16bf;
typedef __attribute__((ext_vector_type(8)))  __bf16 v8bf;
typedef __attribute__((ext_vector_type(4)))  __bf16 v4bf;
typedef __attribute__((ext_vector_type(8)))  float  v8f;

#define B_   8
#define S_   2048
#define D_   1024
#define BM   32
#define BN   32
#define DW   128

// element strides (bf16 / f32 elements)
#define XS_S 1032   // 2064 B per row, 16B aligned
#define YS_S 1032
#define YT_S 40     // 80 B per row, 16B aligned
#define SB_S 33
#define PS_S 40

// byte offsets into dynamic LDS
#define OFF_XS 0
#define OFF_YS 66048
#define OFF_YT 132096
#define OFF_SB 214016
#define OFF_PS 218240
#define OFF_M  220800
#define OFF_L  220928
#define OFF_SC 221056
#define OFF_IV 221184
#define SMEM_BYTES 221312

union AF { v16bf v; v8bf h[2]; };

__device__ __forceinline__ v8f wmma_bf16(v16bf a, v16bf b, v8f c) {
    return __builtin_amdgcn_wmma_f32_16x16x32_bf16(false, a, false, b, (short)0, c, false, false);
}

__device__ __forceinline__ v4bf cvt4(float4 v) {
    v4bf p;
    p[0] = (__bf16)v.x; p[1] = (__bf16)v.y;
    p[2] = (__bf16)v.z; p[3] = (__bf16)v.w;
    return p;
}

__global__ __launch_bounds__(256)
void BasicAttention_fa_kernel(const float* __restrict__ x,
                              const float* __restrict__ y,
                              float* __restrict__ out)
{
    extern __shared__ char smem[];
    __bf16* Xs = (__bf16*)(smem + OFF_XS);
    __bf16* Ys = (__bf16*)(smem + OFF_YS);
    __bf16* Yt = (__bf16*)(smem + OFF_YT);
    float*  Sb = (float*)(smem + OFF_SB);
    __bf16* Ps = (__bf16*)(smem + OFF_PS);
    float*  Mr = (float*)(smem + OFF_M);
    float*  Lr = (float*)(smem + OFF_L);
    float*  Sc = (float*)(smem + OFF_SC);
    float*  Iv = (float*)(smem + OFF_IV);

    const int t    = threadIdx.x;
    const int w    = t >> 5;
    const int lane = t & 31;
    const int m    = lane & 15;   // column/row-within-tile index
    const int h    = lane >> 4;   // lane half
    const int b    = blockIdx.y;
    const int i0   = blockIdx.x * BM;

    // ---------- phase 0: load X tile (fp32 -> bf16 LDS) + passthrough copy ----------
    {
        const int row = t >> 3;
        const int cb  = (t & 7) * 128;
        const float4* xg = (const float4*)(x + ((size_t)b * S_ + (i0 + row)) * D_ + cb);
        float4*       og = (float4*)(out + ((size_t)b * S_ + (i0 + row)) * (2 * D_) + cb);
        __bf16* xrow = Xs + row * XS_S + cb;
        for (int q = 0; q < 32; ++q) {
            float4 v = xg[q];
            og[q] = v;                                  // out[:, :, 0:D] = x
            *(v4bf*)(xrow + q * 4) = cvt4(v);
        }
    }
    if (t < BM) { Mr[t] = -3.0e38f; Lr[t] = 0.0f; }

    v8f O[2][8];
#pragma unroll
    for (int a = 0; a < 2; ++a)
#pragma unroll
        for (int c = 0; c < 8; ++c) O[a][c] = (v8f){0,0,0,0,0,0,0,0};

    __syncthreads();

    const int tile  = w & 3;
    const int mt1   = tile >> 1;   // M tile (0..1) for score matmul
    const int nt1   = tile & 1;    // N tile (0..1) for score matmul
    const int kh    = w >> 2;      // K half (0..1): waves pair up to split D
    const int dbase = w * DW;      // this wave's private D-slice for O

    // Y tile loader mapping: each thread owns a 4(j) x 32(d) block, processed
    // as eight 4x4 micro-blocks so the Yt (transposed) writes are 8B stores.
    const int jt = (t >> 5) * 4;       // 0,4,..,28
    const int dt = (t & 31) * 32;      // 0,32,..,992

    for (int j0 = 0; j0 < S_; j0 += BN) {
        // ---------- phase A: load Y tile -> Ys (row major) + Yt (transposed); zero Sb ----------
        {
            const float* yg0 = y + ((size_t)b * S_ + (j0 + jt + 0)) * D_ + dt;
            const float* yg1 = yg0 + D_;
            const float* yg2 = yg1 + D_;
            const float* yg3 = yg2 + D_;
            // prefetch next j-tile's slice of y into cache while we work
            if (j0 + BN < S_) {
                __builtin_prefetch(yg0 + (size_t)BN * D_, 0, 1);
                __builtin_prefetch(yg2 + (size_t)BN * D_, 0, 1);
            }
            for (int q = 0; q < 8; ++q) {
                const int d = q * 4;
                float4 r0 = *(const float4*)(yg0 + d);
                float4 r1 = *(const float4*)(yg1 + d);
                float4 r2 = *(const float4*)(yg2 + d);
                float4 r3 = *(const float4*)(yg3 + d);
                v4bf p0 = cvt4(r0), p1 = cvt4(r1), p2 = cvt4(r2), p3 = cvt4(r3);
                *(v4bf*)(Ys + (jt + 0) * YS_S + dt + d) = p0;
                *(v4bf*)(Ys + (jt + 1) * YS_S + dt + d) = p1;
                *(v4bf*)(Ys + (jt + 2) * YS_S + dt + d) = p2;
                *(v4bf*)(Ys + (jt + 3) * YS_S + dt + d) = p3;
                // 4x4 in-register transpose -> 8B stores into Yt rows
#pragma unroll
                for (int e = 0; e < 4; ++e) {
                    v4bf c;
                    c[0] = p0[e]; c[1] = p1[e]; c[2] = p2[e]; c[3] = p3[e];
                    *(v4bf*)(Yt + (dt + d + e) * YT_S + jt) = c;
                }
            }
            float* sb = Sb + (t >> 3) * SB_S + (t & 7) * 4;
            sb[0] = 0.f; sb[1] = 0.f; sb[2] = 0.f; sb[3] = 0.f;
        }
        __syncthreads();

        // ---------- phase B: S(partial) = X * Y^T over this wave's K half ----------
        {
            v8f acc = (v8f){0,0,0,0,0,0,0,0};
            const __bf16* xr = Xs + (mt1 * 16 + m) * XS_S;   // A: per-lane row
            const __bf16* yr = Ys + (nt1 * 16 + m) * YS_S;   // B: per-lane column (= y row)
            const int k0 = kh * 512;
            for (int kk = k0; kk < k0 + 512; kk += 32) {
                AF a, bb;
                a.h[0]  = *(const v8bf*)(xr + kk + 8 * h);        // K = 8h .. 8h+7
                a.h[1]  = *(const v8bf*)(xr + kk + 8 * h + 16);   // K = 16+8h ..
                bb.h[0] = *(const v8bf*)(yr + kk + 8 * h);        // K = 8h .. 8h+7
                bb.h[1] = *(const v8bf*)(yr + kk + 8 * h + 16);   // K = 16+8h ..
                acc = wmma_bf16(a.v, bb.v, acc);
            }
#pragma unroll
            for (int r = 0; r < 8; ++r) {
                const int rowS = mt1 * 16 + 8 * h + r;
                atomicAdd(&Sb[rowS * SB_S + nt1 * 16 + m], acc[r]);
            }
        }
        __syncthreads();

        // ---------- phase C: online softmax on Sb rows -> Ps, update running stats ----------
        {
            const int r  = t >> 3;
            const int c0 = (t & 7) * 4;
            float* sp = Sb + r * SB_S + c0;
            float s0 = sp[0], s1 = sp[1], s2 = sp[2], s3 = sp[3];
            float mx = fmaxf(fmaxf(s0, s1), fmaxf(s2, s3));
            mx = fmaxf(mx, __shfl_xor(mx, 1));
            mx = fmaxf(mx, __shfl_xor(mx, 2));
            mx = fmaxf(mx, __shfl_xor(mx, 4));
            const float mold = Mr[r];
            const float mnew = fmaxf(mold, mx);
            float p0 = __expf(s0 - mnew), p1 = __expf(s1 - mnew);
            float p2 = __expf(s2 - mnew), p3 = __expf(s3 - mnew);
            v4bf pv;
            pv[0] = (__bf16)p0; pv[1] = (__bf16)p1;
            pv[2] = (__bf16)p2; pv[3] = (__bf16)p3;
            *(v4bf*)(Ps + r * PS_S + c0) = pv;
            float ls = p0 + p1 + p2 + p3;
            ls += __shfl_xor(ls, 1);
            ls += __shfl_xor(ls, 2);
            ls += __shfl_xor(ls, 4);
            if ((t & 7) == 0) {
                const float sc = __expf(mold - mnew);
                Lr[r] = Lr[r] * sc + ls;
                Mr[r] = mnew;
                Sc[r] = sc;
            }
        }
        __syncthreads();

        // ---------- phase D: rescale O by softmax correction, O += P * Y (this D slice) ----------
        {
            AF pa[2];
#pragma unroll
            for (int mt = 0; mt < 2; ++mt) {
                const __bf16* pr = Ps + (mt * 16 + m) * PS_S + 8 * h;
                pa[mt].h[0] = *(const v8bf*)pr;
                pa[mt].h[1] = *(const v8bf*)(pr + 16);
                float scv[8];
#pragma unroll
                for (int r = 0; r < 8; ++r) scv[r] = Sc[mt * 16 + 8 * h + r];
#pragma unroll
                for (int nt = 0; nt < 8; ++nt)
#pragma unroll
                    for (int r = 0; r < 8; ++r) O[mt][nt][r] *= scv[r];
            }
#pragma unroll
            for (int nt = 0; nt < 8; ++nt) {
                AF bb;
                const __bf16* yc = Yt + (dbase + nt * 16 + m) * YT_S + 8 * h;
                bb.h[0] = *(const v8bf*)yc;        // K (= j) 8h .. 8h+7
                bb.h[1] = *(const v8bf*)(yc + 16); // K 16+8h ..
                O[0][nt] = wmma_bf16(pa[0].v, bb.v, O[0][nt]);
                O[1][nt] = wmma_bf16(pa[1].v, bb.v, O[1][nt]);
            }
        }
        __syncthreads();
    }

    if (t < BM) Iv[t] = 1.0f / Lr[t];
    __syncthreads();

    // ---------- epilogue: out[:, :, D:2D] = O / l ----------
#pragma unroll
    for (int mt = 0; mt < 2; ++mt) {
        float iv[8];
#pragma unroll
        for (int r = 0; r < 8; ++r) iv[r] = Iv[mt * 16 + 8 * h + r];
#pragma unroll
        for (int nt = 0; nt < 8; ++nt) {
            const int col = D_ + dbase + nt * 16 + m;
#pragma unroll
            for (int r = 0; r < 8; ++r) {
                const int row = i0 + mt * 16 + 8 * h + r;
                out[((size_t)b * S_ + row) * (2 * D_) + col] = O[mt][nt][r] * iv[r];
            }
        }
    }
}

extern "C" void kernel_launch(void* const* d_in, const int* in_sizes, int n_in,
                              void* d_out, int out_size, void* d_ws, size_t ws_size,
                              hipStream_t stream) {
    const float* x = (const float*)d_in[0];
    const float* y = (const float*)d_in[1];
    float* out = (float*)d_out;
    dim3 grid(S_ / BM, B_);
    BasicAttention_fa_kernel<<<grid, 256, SMEM_BYTES, stream>>>(x, y, out);
}
